// Attention_21895743275483
// MI455X (gfx1250) — matrix operand order repaired and checked by reading, not compiled, not run on hardware
//
#include <hip/hip_runtime.h>
#include <stdint.h>

// ---------------- problem constants (match reference) ----------------
#define B_  2
#define N_  32
#define T_  2048
#define S_  2048
#define H_  128

#define BLK_T 64        // query rows per workgroup (4 waves x 16)
#define BLK_S 64        // key/value columns per main-loop iteration
#define NWAVE 4

// padded LDS strides (bf16 elements):
//  - fragment reads are 16B-aligned b128s, row bank-step = stride/2 dwords
//  - QSTR: step 68 mod 64 = 4  -> 16 distinct banks
//  - VSTR/PSTR: step 36, gcd(36,64)=4 -> 16 distinct banks
#define QSTR 136        // 128 + 8  (Q and K tiles, row major)
#define VSTR 72         // 64 + 8   (V tile, transposed: [h][s])
#define PSTR 72         // 64 + 8   (per-wave P scratch)

typedef __bf16 bf16_t;
typedef __attribute__((ext_vector_type(16))) __bf16 v16bf;
typedef __attribute__((ext_vector_type(8)))  float  v8f;

union V16U { v16bf v; uint4 u[2]; };
union V8U  { bf16_t h[8]; uint4 u; };
union P2U  { bf16_t h[2]; uint32_t u; };

__device__ __forceinline__ uint4 lds_ld16(const bf16_t* p) {
    return *(const uint4*)p;
}

// convert 8 consecutive f32 -> 8 bf16, one 16B LDS write
__device__ __forceinline__ void cvt_store8(bf16_t* dst, float4 a, float4 b) {
    V8U t;
    t.h[0] = (bf16_t)a.x; t.h[1] = (bf16_t)a.y; t.h[2] = (bf16_t)a.z; t.h[3] = (bf16_t)a.w;
    t.h[4] = (bf16_t)b.x; t.h[5] = (bf16_t)b.y; t.h[6] = (bf16_t)b.z; t.h[7] = (bf16_t)b.w;
    *(uint4*)dst = t.u;
}

// pack two f32 -> two bf16, one 4B LDS write
__device__ __forceinline__ void cvt_store2(bf16_t* dst, float a, float b) {
    P2U t;
    t.h[0] = (bf16_t)a; t.h[1] = (bf16_t)b;
    *(uint32_t*)dst = t.u;
}

__global__ __launch_bounds__(NWAVE * 32) void
fattn_fwd_kernel(const float* __restrict__ q, const float* __restrict__ k,
                 const float* __restrict__ v, const unsigned char* __restrict__ mask,
                 float* __restrict__ out)
{
    __shared__ bf16_t qlds[BLK_T * QSTR];        // 17.0 KB
    __shared__ bf16_t klds[BLK_S * QSTR];        // 17.0 KB
    __shared__ bf16_t vtlds[H_ * VSTR];          // 18.0 KB  (transposed V: [h][s])
    __shared__ bf16_t plds[NWAVE * 16 * PSTR];   //  9.0 KB  (per-wave P scratch)

    const int tid  = threadIdx.x;
    const int wave = tid >> 5;
    const int lane = tid & 31;
    const int m    = lane & 15;      // fragment row / column index
    const int hi   = lane >> 4;      // lane half (0/1)

    const int bn = blockIdx.y;       // flattened (b, n)
    const int b  = bn / N_;
    const int t0 = blockIdx.x * BLK_T;

    const float* qb = q + ((size_t)bn * T_ + t0) * H_;
    const float* kb = k + (size_t)bn * S_ * H_;
    const float* vb = v + (size_t)bn * S_ * H_;
    const unsigned char* mb = mask + ((size_t)b * T_ + t0) * S_;

    // ---------------- stage Q block: 64x128 f32 -> bf16 LDS ----------------
    {
        const int row = tid >> 1;
        const int c0  = (tid & 1) * 64;
        const float* src = qb + (size_t)row * H_ + c0;
        bf16_t* dst = &qlds[row * QSTR + c0];
        #pragma unroll
        for (int i = 0; i < 8; ++i) {
            float4 a = ((const float4*)src)[2 * i];
            float4 c = ((const float4*)src)[2 * i + 1];
            cvt_store8(dst + 8 * i, a, c);
        }
    }
    __syncthreads();

    // ---------------- per-wave Q A-fragments (16x32 bf16, 4 k-chunks) -------
    v16bf qf[4];
    {
        const bf16_t* qrow = &qlds[(wave * 16 + m) * QSTR];
        #pragma unroll
        for (int c = 0; c < 4; ++c) {
            V16U t;
            t.u[0] = lds_ld16(qrow + 32 * c + hi * 16);
            t.u[1] = lds_ld16(qrow + 32 * c + hi * 16 + 8);
            qf[c] = t.v;
        }
    }

    // ---------------- accumulators + softmax state -------------------------
    v8f acc[8];
    #pragma unroll
    for (int jh = 0; jh < 8; ++jh) acc[jh] = v8f{};
    float mrow[8], lrow[8];
    #pragma unroll
    for (int r = 0; r < 8; ++r) { mrow[r] = -__builtin_inff(); lrow[r] = 0.f; }

    const float QSCALE   = 0.08838834764831845f * 1.4426950408889634f; // 1/sqrt(128)*log2(e)
    const float MASK_NEG = -1.0e30f;   // finite: fully-masked row -> uniform softmax

    bf16_t* pw = &plds[wave * 16 * PSTR];

    for (int s0 = 0; s0 < S_; s0 += BLK_S) {
        __syncthreads();   // klds / vtlds reuse barrier

        // prefetch next K/V tile into cache (global_prefetch_b8)
        if (s0 + BLK_S < S_) {
            const int prow = tid >> 1;
            const int pc0  = (tid & 1) * 64;
            __builtin_prefetch(kb + (size_t)(s0 + BLK_S + prow) * H_ + pc0, 0, 1);
            __builtin_prefetch(vb + (size_t)(s0 + BLK_S + prow) * H_ + pc0, 0, 1);
        }

        // ---- cooperative K tile (64x128) -> klds row-major bf16 ----
        {
            const int row = tid >> 1;
            const int c0  = (tid & 1) * 64;
            const float* src = kb + (size_t)(s0 + row) * H_ + c0;
            bf16_t* dst = &klds[row * QSTR + c0];
            #pragma unroll
            for (int i = 0; i < 8; ++i) {
                float4 a = ((const float4*)src)[2 * i];
                float4 c = ((const float4*)src)[2 * i + 1];
                cvt_store8(dst + 8 * i, a, c);
            }
        }
        // ---- cooperative V tile (64x128) -> vtlds transposed [h][s] ----
        // thread owns an (s, s+1) pair at a 16-wide h chunk; packed b32 stores
        // lane->s-pair mapping puts the 32 lanes on 32 consecutive dwords (distinct banks).
        {
            #pragma unroll
            for (int a = 0; a < 2; ++a) {
                const int sp = lane;                    // s pair index 0..31
                const int hc = (tid >> 5) + 4 * a;      // h chunk 0..7
                const int s  = 2 * sp;
                const float* r0 = vb + (size_t)(s0 + s) * H_ + hc * 16;
                const float* r1 = r0 + H_;
                bf16_t* dst = &vtlds[(hc * 16) * VSTR + s];
                #pragma unroll
                for (int i = 0; i < 4; ++i) {
                    float4 x = ((const float4*)r0)[i];
                    float4 y = ((const float4*)r1)[i];
                    cvt_store2(dst + (4 * i + 0) * VSTR, x.x, y.x);
                    cvt_store2(dst + (4 * i + 1) * VSTR, x.y, y.y);
                    cvt_store2(dst + (4 * i + 2) * VSTR, x.z, y.z);
                    cvt_store2(dst + (4 * i + 3) * VSTR, x.w, y.w);
                }
            }
        }
        __syncthreads();

        // ---- scores S = Q @ K^T : four 16-col tiles, 4 k-chunks each ----
        // software-pipelined: fragments for chunk c+1 load while chunk c WMMAs run
        v8f sa[4];
        #pragma unroll
        for (int j = 0; j < 4; ++j) sa[j] = v8f{};
        V16U kf[4];
        #pragma unroll
        for (int j = 0; j < 4; ++j) {
            const bf16_t* kr = &klds[(j * 16 + m) * QSTR + hi * 16];
            kf[j].u[0] = lds_ld16(kr);
            kf[j].u[1] = lds_ld16(kr + 8);
        }
        #pragma unroll
        for (int c = 0; c < 4; ++c) {
            V16U kn[4];
            if (c < 3) {
                #pragma unroll
                for (int j = 0; j < 4; ++j) {
                    const bf16_t* kr = &klds[(j * 16 + m) * QSTR + 32 * (c + 1) + hi * 16];
                    kn[j].u[0] = lds_ld16(kr);
                    kn[j].u[1] = lds_ld16(kr + 8);
                }
            }
            #pragma unroll
            for (int j = 0; j < 4; ++j)
                sa[j] = __builtin_amdgcn_wmma_f32_16x16x32_bf16(
                            false, qf[c], false, kf[j].v, (short)0, sa[j], false, false);
            if (c < 3) {
                #pragma unroll
                for (int j = 0; j < 4; ++j) kf[j] = kn[j];
            }
        }

        // ---- scale + mask (C layout: VGPR r -> row r + 8*hi, col = m + 16j) ----
        float sc[4][8];
        const unsigned char* mrp = mb + (size_t)(wave * 16 + 8 * hi) * S_ + s0 + m;
        #pragma unroll
        for (int j = 0; j < 4; ++j) {
            #pragma unroll
            for (int r = 0; r < 8; ++r) {
                unsigned char mk = mrp[(size_t)r * S_ + 16 * j];
                float x = sa[j][r] * QSCALE;
                sc[j][r] = mk ? x : MASK_NEG;
            }
        }

        // ---- online softmax (row stats via 16-lane shfl reductions) ----
        float tmax[8];
        #pragma unroll
        for (int r = 0; r < 8; ++r)
            tmax[r] = fmaxf(fmaxf(sc[0][r], sc[1][r]), fmaxf(sc[2][r], sc[3][r]));
        #pragma unroll
        for (int off = 8; off >= 1; off >>= 1) {
            #pragma unroll
            for (int r = 0; r < 8; ++r)
                tmax[r] = fmaxf(tmax[r], __shfl_xor(tmax[r], off, 16));
        }

        float mnew[8], alpha[8];
        #pragma unroll
        for (int r = 0; r < 8; ++r) {
            mnew[r]  = fmaxf(mrow[r], tmax[r]);
            alpha[r] = __builtin_amdgcn_exp2f(mrow[r] - mnew[r]);
            mrow[r]  = mnew[r];
        }

        float psum[8];
        #pragma unroll
        for (int r = 0; r < 8; ++r) psum[r] = 0.f;
        #pragma unroll
        for (int j = 0; j < 4; ++j) {
            #pragma unroll
            for (int r = 0; r < 8; ++r) {
                float p = __builtin_amdgcn_exp2f(sc[j][r] - mnew[r]);
                psum[r] += p;
                pw[(r + 8 * hi) * PSTR + 16 * j + m] = (bf16_t)p;   // C -> row-major
            }
        }
        #pragma unroll
        for (int off = 8; off >= 1; off >>= 1) {
            #pragma unroll
            for (int r = 0; r < 8; ++r)
                psum[r] += __shfl_xor(psum[r], off, 16);
        }
        #pragma unroll
        for (int r = 0; r < 8; ++r)
            lrow[r] = lrow[r] * alpha[r] + psum[r];

        // rescale output accumulators
        #pragma unroll
        for (int jh = 0; jh < 8; ++jh) {
            #pragma unroll
            for (int r = 0; r < 8; ++r)
                acc[jh][r] *= alpha[r];
        }

        // ---- O += P @ V : two 32-deep k-chunks, 8 h-tiles of 16 ----
        #pragma unroll
        for (int kc = 0; kc < 2; ++kc) {
            V16U pa;
            {
                const bf16_t* pr = &pw[m * PSTR + 32 * kc + hi * 16];
                pa.u[0] = lds_ld16(pr);
                pa.u[1] = lds_ld16(pr + 8);
            }
            #pragma unroll
            for (int g = 0; g < 2; ++g) {
                V16U vf[4];
                #pragma unroll
                for (int jj = 0; jj < 4; ++jj) {
                    const int jh = g * 4 + jj;
                    const bf16_t* vr = &vtlds[(jh * 16 + m) * VSTR + 32 * kc + hi * 16];
                    vf[jj].u[0] = lds_ld16(vr);
                    vf[jj].u[1] = lds_ld16(vr + 8);
                }
                #pragma unroll
                for (int jj = 0; jj < 4; ++jj) {
                    const int jh = g * 4 + jj;
                    acc[jh] = __builtin_amdgcn_wmma_f32_16x16x32_bf16(
                                  false, pa.v, false, vf[jj].v, (short)0, acc[jh], false, false);
                }
            }
        }
    }

    // ---------------- epilogue: normalize + store --------------------------
    #pragma unroll
    for (int r = 0; r < 8; ++r) lrow[r] = 1.f / lrow[r];

    float* ob = out + ((size_t)bn * T_ + t0 + wave * 16) * H_;
    #pragma unroll
    for (int jh = 0; jh < 8; ++jh) {
        #pragma unroll
        for (int r = 0; r < 8; ++r)
            ob[(size_t)(r + 8 * hi) * H_ + 16 * jh + m] = acc[jh][r] * lrow[r];
    }
}

extern "C" void kernel_launch(void* const* d_in, const int* in_sizes, int n_in,
                              void* d_out, int out_size, void* d_ws, size_t ws_size,
                              hipStream_t stream) {
    (void)in_sizes; (void)n_in; (void)out_size; (void)d_ws; (void)ws_size;
    const float* q = (const float*)d_in[0];
    const float* k = (const float*)d_in[1];
    const float* v = (const float*)d_in[2];
    const unsigned char* mask = (const unsigned char*)d_in[3];
    float* out = (float*)d_out;

    dim3 grid(T_ / BLK_T, B_ * N_);
    dim3 block(NWAVE * 32);
    fattn_fwd_kernel<<<grid, block, 0, stream>>>(q, k, v, mask, out);
}
